// TPMSLimbRouter_15814069584146
// MI455X (gfx1250) — compile-verified, matrix operand order checked by reading, not run on hardware
//
#include <hip/hip_runtime.h>
#include <hip/hip_bf16.h>

// Shapes from the reference
#define HD    1024          // H
#define LIMBS 8
#define BD    4
#define SD    2048
#define MROWS (BD * SD * LIMBS)   // 65536
#define MTILE 64                  // rows per workgroup (8 limb groups)
#define LDT   1032                // padded LDS row stride (bf16 elements), mult of 8

typedef __attribute__((ext_vector_type(16))) __bf16          v16bf;
typedef __attribute__((ext_vector_type(8)))  float           v8f;
typedef __attribute__((ext_vector_type(8)))  unsigned short  v8u16;

union BFrag { v16bf bf; v8u16 h[2]; };

__device__ __forceinline__ unsigned short f2bf(float f) {
    unsigned u = __float_as_uint(f);
    unsigned r = (u + 0x7FFFu + ((u >> 16) & 1u)) >> 16;   // round-nearest-even
    return (unsigned short)r;
}
__device__ __forceinline__ float bf2f(unsigned short h) {
    return __uint_as_float(((unsigned)h) << 16);
}

// A-fragment (16x32 bf16, M x K) from an LDS tile. Per ISA 7.12.2:
// lanes 0-15: M=lane, K = {0..7, 16..23}; lanes 16-31: M=lane-16, K = {8..15, 24..31}
__device__ __forceinline__ v16bf load_a_lds(const unsigned short* tile, int m0, int k0, int lane) {
    const int m  = m0 + (lane & 15);
    const int kh = (lane >> 4) << 3;               // 0 or 8
    const unsigned short* p = tile + m * LDT + k0 + kh;
    BFrag f;
    f.h[0] = *(const v8u16*)(p);                   // K = k0+kh .. +7
    f.h[1] = *(const v8u16*)(p + 16);              // K = k0+kh+16 .. +23
    return f.bf;
}

// B-fragment (32x16 bf16, K x N) for C = A*W^T with W row-major [N][K]:
// same per-lane layout as A with lane indexing N, K-contiguous loads.
__device__ __forceinline__ v16bf load_b_w(const unsigned short* __restrict__ w, int ldw,
                                          int n0, int k0, int lane) {
    const int n  = n0 + (lane & 15);
    const int kh = (lane >> 4) << 3;
    const unsigned short* p = w + (size_t)n * ldw + k0 + kh;
    BFrag f;
    f.h[0] = *(const v8u16*)(p);
    f.h[1] = *(const v8u16*)(p + 16);
    return f.bf;
}

__device__ __forceinline__ v8f wmma_bf16(v16bf a, v16bf b, v8f c) {
    return __builtin_amdgcn_wmma_f32_16x16x32_bf16(false, a, false, b, (short)0, c, false, false);
}

// ---------------- prologue: f32 -> bf16 weight conversion ----------------
__global__ void cvt_bf16_kernel(const float* __restrict__ src, unsigned short* __restrict__ dst, int n) {
    int i = blockIdx.x * blockDim.x + threadIdx.x;
    if (i < n) dst[i] = f2bf(src[i]);
}

// ---------------- prologue: 8x8 routing matrix ----------------
__global__ void routing_kernel(const float* __restrict__ pos, const float* __restrict__ thr,
                               float* __restrict__ R) {
    int i = threadIdx.x;
    if (i >= LIMBS) return;
    float t  = thr[0];
    float px = pos[i * 3 + 0], py = pos[i * 3 + 1], pz = pos[i * 3 + 2];
    float row[LIMBS];
    float sum = 0.f;
    #pragma unroll
    for (int j = 0; j < LIMBS; ++j) {
        float mx = 0.5f * (px + pos[j * 3 + 0]);
        float my = 0.5f * (py + pos[j * 3 + 1]);
        float mz = 0.5f * (pz + pos[j * 3 + 2]);
        float f  = __sinf(mx) * __cosf(my) + __sinf(my) * __cosf(mz) + __sinf(mz) * __cosf(mx) - t;
        float r  = 1.f / (1.f + __expf(-2.f * f));
        if (j == i) r = 1.f;                        // diagonal forced to 1
        row[j] = r;
        sum += r;
    }
    float inv = 1.f / (sum + 1e-6f);
    #pragma unroll
    for (int j = 0; j < LIMBS; ++j) R[i * LIMBS + j] = row[j] * inv;
}

// ---------------- fused main kernel ----------------
// Per workgroup: 64 rows; LDS holds x-tile and routed-tile as bf16 (~258 KB of 320 KB).
// Wave mapping: each wave covers ALL 64 rows x 32 cols, so every B (weight) fragment
// is fetched exactly once per workgroup -> minimal L2 weight traffic for this fusion.
__global__ __launch_bounds__(256)
void tpms_fused_kernel(const float* __restrict__ x,
                       const unsigned short* __restrict__ wr_bf,   // [H][H]  bf16
                       const float* __restrict__ b_route,          // [H]
                       const unsigned short* __restrict__ wg_bf,   // [H][2H] bf16
                       const float* __restrict__ b_gate,           // [H]
                       const float* __restrict__ Rmat,             // [8][8]
                       float* __restrict__ out) {
    extern __shared__ char smem[];
    unsigned short* xs   = (unsigned short*)smem;            // [MTILE][LDT] x tile (bf16)
    unsigned short* rs   = xs + MTILE * LDT;                 // [MTILE][LDT] routed tile (bf16)
    float*          ldsR = (float*)(rs + MTILE * LDT);       // [64]

    const int    tid     = threadIdx.x;
    const int    lane    = tid & 31;
    const int    wave    = tid >> 5;                         // 0..7
    const size_t rowBase = (size_t)blockIdx.x * MTILE;

    if (tid < 64) ldsR[tid] = Rmat[tid];

    // ---- Phase 0: stage x tile into LDS as bf16 (vectorized, coalesced) ----
    for (int i = tid; i < MTILE * HD / 4; i += 256) {
        int idx = i * 4;
        int r   = idx >> 10;                                 // / HD
        int c   = idx & (HD - 1);
        const float4 v = *(const float4*)(x + (rowBase + r) * HD + c);
        ushort4 h;
        h.x = f2bf(v.x); h.y = f2bf(v.y); h.z = f2bf(v.z); h.w = f2bf(v.w);
        *(ushort4*)(xs + r * LDT + c) = h;
    }
    __syncthreads();

    // routing matrix into registers (uniform across lanes)
    float Rr[64];
    #pragma unroll
    for (int i = 0; i < 64; ++i) Rr[i] = ldsR[i];

    const int colA = (lane & 15);
    const int rhh  = (lane >> 4) << 3;                       // C-layout: VGPR r -> row m0 + rhh + r

    // ---- Phase 1: projected = x @ Wr^T + b, then 8x8 limb mix in VGPRs ----
    // 8 waves x 32 cols = 256 cols per outer iteration -> 4 iterations over H.
    for (int nb = 0; nb < HD; nb += 256) {
        const int n0 = nb + wave * 32;
        v8f acc[4][2];
        #pragma unroll
        for (int rb = 0; rb < 4; ++rb) { acc[rb][0] = v8f{}; acc[rb][1] = v8f{}; }

        for (int k0 = 0; k0 < HD; k0 += 32) {
            v16bf b0 = load_b_w(wr_bf, HD, n0,      k0, lane);
            v16bf b1 = load_b_w(wr_bf, HD, n0 + 16, k0, lane);
            #pragma unroll
            for (int rb = 0; rb < 4; ++rb) {
                v16bf a = load_a_lds(xs, rb * 16, k0, lane);
                acc[rb][0] = wmma_bf16(a, b0, acc[rb][0]);
                acc[rb][1] = wmma_bf16(a, b1, acc[rb][1]);
            }
        }

        const float bias0 = b_route[n0 + colA];
        const float bias1 = b_route[n0 + 16 + colA];
        #pragma unroll
        for (int rb = 0; rb < 4; ++rb) {
            #pragma unroll
            for (int r = 0; r < 8; ++r) { acc[rb][0][r] += bias0; acc[rb][1][r] += bias1; }

            // limb mix: accumulator VGPR index == limb index within group
            v8f m0, m1;
            #pragma unroll
            for (int r = 0; r < 8; ++r) {
                float s0 = 0.f, s1 = 0.f;
                #pragma unroll
                for (int j = 0; j < 8; ++j) {
                    const float rv = Rr[r * 8 + j];
                    s0 = __builtin_fmaf(rv, acc[rb][0][j], s0);
                    s1 = __builtin_fmaf(rv, acc[rb][1][j], s1);
                }
                m0[r] = s0; m1[r] = s1;
            }
            const int rowh = rb * 16 + rhh;
            #pragma unroll
            for (int r = 0; r < 8; ++r) {
                rs[(rowh + r) * LDT + n0 + colA]      = f2bf(m0[r]);
                rs[(rowh + r) * LDT + n0 + 16 + colA] = f2bf(m1[r]);
            }
        }
    }
    __syncthreads();

    // ---- Phase 2: gate = sigmoid([x | routed] @ Wg^T + b); blend; store ----
    for (int nb = 0; nb < HD; nb += 256) {
        const int n0 = nb + wave * 32;
        v8f acc[4][2];
        #pragma unroll
        for (int rb = 0; rb < 4; ++rb) { acc[rb][0] = v8f{}; acc[rb][1] = v8f{}; }

        for (int k0 = 0; k0 < HD; k0 += 32) {                // K < H : x half
            v16bf b0 = load_b_w(wg_bf, 2 * HD, n0,      k0, lane);
            v16bf b1 = load_b_w(wg_bf, 2 * HD, n0 + 16, k0, lane);
            #pragma unroll
            for (int rb = 0; rb < 4; ++rb) {
                v16bf a = load_a_lds(xs, rb * 16, k0, lane);
                acc[rb][0] = wmma_bf16(a, b0, acc[rb][0]);
                acc[rb][1] = wmma_bf16(a, b1, acc[rb][1]);
            }
        }
        for (int k0 = 0; k0 < HD; k0 += 32) {                // K >= H : routed half
            v16bf b0 = load_b_w(wg_bf, 2 * HD, n0,      HD + k0, lane);
            v16bf b1 = load_b_w(wg_bf, 2 * HD, n0 + 16, HD + k0, lane);
            #pragma unroll
            for (int rb = 0; rb < 4; ++rb) {
                v16bf a = load_a_lds(rs, rb * 16, k0, lane);
                acc[rb][0] = wmma_bf16(a, b0, acc[rb][0]);
                acc[rb][1] = wmma_bf16(a, b1, acc[rb][1]);
            }
        }

        const int cA  = n0 + colA;
        const int cB  = cA + 16;
        const float bg0 = b_gate[cA];
        const float bg1 = b_gate[cB];
        #pragma unroll
        for (int rb = 0; rb < 4; ++rb) {
            const int rowh = rb * 16 + rhh;
            #pragma unroll
            for (int r = 0; r < 8; ++r) {
                const int row = rowh + r;
                const float rt0 = bf2f(rs[row * LDT + cA]);
                const float rt1 = bf2f(rs[row * LDT + cB]);
                const float xv0 = bf2f(xs[row * LDT + cA]);
                const float xv1 = bf2f(xs[row * LDT + cB]);
                const float g0  = 1.f / (1.f + __expf(-(acc[rb][0][r] + bg0)));
                const float g1  = 1.f / (1.f + __expf(-(acc[rb][1][r] + bg1)));
                const size_t o  = (rowBase + row) * HD;
                out[o + cA] = g0 * rt0 + (1.f - g0) * xv0;
                out[o + cB] = g1 * rt1 + (1.f - g1) * xv1;
            }
        }
    }
}

extern "C" void kernel_launch(void* const* d_in, const int* in_sizes, int n_in,
                              void* d_out, int out_size, void* d_ws, size_t ws_size,
                              hipStream_t stream) {
    const float* x   = (const float*)d_in[0];   // [B,S,L,H]
    const float* pos = (const float*)d_in[1];   // [8,3]
    const float* thr = (const float*)d_in[2];   // scalar
    const float* Wr  = (const float*)d_in[3];   // [H,H]
    const float* br  = (const float*)d_in[4];   // [H]
    const float* Wg  = (const float*)d_in[5];   // [H,2H]
    const float* bg  = (const float*)d_in[6];   // [H]
    float* out = (float*)d_out;

    // workspace: bf16 Wr (2MB) | bf16 Wg (4MB) | routing matrix (256B)
    unsigned short* wr_bf = (unsigned short*)d_ws;
    unsigned short* wg_bf = wr_bf + (size_t)HD * HD;
    float*          Rmat  = (float*)(wg_bf + (size_t)HD * 2 * HD);

    const int nWr = HD * HD;
    const int nWg = HD * 2 * HD;
    cvt_bf16_kernel<<<(nWr + 255) / 256, 256, 0, stream>>>(Wr, wr_bf, nWr);
    cvt_bf16_kernel<<<(nWg + 255) / 256, 256, 0, stream>>>(Wg, wg_bf, nWg);
    routing_kernel<<<1, 32, 0, stream>>>(pos, thr, Rmat);

    const size_t smem = (size_t)(2 * MTILE * LDT) * sizeof(unsigned short) + 64 * sizeof(float);
    tpms_fused_kernel<<<MROWS / MTILE, 256, smem, stream>>>(x, wr_bf, br, wg_bf, bg, Rmat, out);
}
